// STTransformerLayer_51608327029325
// MI455X (gfx1250) — compile-verified, matrix operand order checked by reading, not run on hardware
//
#include <hip/hip_runtime.h>
#include <hip/hip_bf16.h>
#include <math.h>

// ---------------------------------------------------------------------------
// STTransformerLayer for MI455X (gfx1250): bf16-native WMMA pipeline.
//   B=16 T=120 J=24 D=512 H=8 dh=64 FF=2048 M=77
// d_out layout (float32): x | sa_w | ta_w | ca_w
// GEMM: 4-wave 64x64 blocks, async global->LDS staging (ASYNCcnt), WMMA bf16.
// ---------------------------------------------------------------------------

typedef __attribute__((ext_vector_type(16))) __bf16 v16bf;
typedef __attribute__((ext_vector_type(8)))  __bf16 v8bf;
typedef __attribute__((ext_vector_type(8)))  float  v8f;
typedef __attribute__((ext_vector_type(4)))  int    v4i;

#define B_  16
#define T_  120
#define J_  24
#define D_  512
#define H_  8
#define DH_ 64
#define FF_ 2048
#define M_  77
#define NTOK (B_*T_*J_)          /* 46080 */
#define TJ_  (T_*J_)             /* 2880  */

#define FLAG_GELU 1
#define FLAG_BF16OUT 2

#if defined(__AMDGCN__) && __has_builtin(__builtin_amdgcn_global_load_async_to_lds_b128)
#define USE_ASYNC_LDS 1
typedef __attribute__((address_space(1))) v4i* gv4i_p;   // global v4i*
typedef __attribute__((address_space(3))) v4i* lv4i_p;   // LDS v4i*
#else
#define USE_ASYNC_LDS 0
#endif

// k-offset inside a 16-bit A/B WMMA fragment (ISA 7.12.2, 16-bit A 16x32):
//   e in 0..7  -> k =  e      + 8*half
//   e in 8..15 -> k = (e-8)+16 + 8*half
__device__ __forceinline__ int koff(int e, int half) {
    int v = e >> 1, j = e & 1;
    int base = (v < 4) ? (2 * v) : (8 + 2 * v);
    return base + 8 * half + j;
}

// load one 16-element bf16 fragment: two contiguous 16B chunks (global or LDS)
__device__ __forceinline__ v16bf ldfrag(const __bf16* p, int half) {
    union { v16bf v; v8bf h[2]; } u;
    u.h[0] = *reinterpret_cast<const v8bf*>(p + 8 * half);
    u.h[1] = *reinterpret_cast<const v8bf*>(p + 16 + 8 * half);
    return u.v;
}

// 16-byte global -> LDS copy (async on CDNA5, fallback: through VGPRs)
__device__ __forceinline__ void copy16(const __bf16* g, __bf16* l) {
#if USE_ASYNC_LDS
    __builtin_amdgcn_global_load_async_to_lds_b128((gv4i_p)g, (lv4i_p)l, 0, 0);
#else
    *reinterpret_cast<v8bf*>(l) = *reinterpret_cast<const v8bf*>(g);
#endif
}

__device__ __forceinline__ void async_wait_all() {
#if USE_ASYNC_LDS
#if __has_builtin(__builtin_amdgcn_s_wait_asynccnt)
    __builtin_amdgcn_s_wait_asynccnt(0);
#else
    asm volatile("s_wait_asynccnt 0x0" ::: "memory");
#endif
#endif
}

// ------------------------- f32 -> bf16 conversion --------------------------
__global__ void cvt_bf16(const float* __restrict__ src, __bf16* __restrict__ dst,
                         size_t n) {
    size_t i = (size_t)blockIdx.x * blockDim.x + threadIdx.x;
    size_t stride = (size_t)gridDim.x * blockDim.x;
    for (; i < n; i += stride) dst[i] = (__bf16)src[i];
}

// ------------------------------- LayerNorm ---------------------------------
// one wave per row of 512; bf16 output (feeds WMMA GEMMs)
__global__ void ln_kernel(const float* __restrict__ x,
                          const float* __restrict__ g,
                          const float* __restrict__ b,
                          __bf16* __restrict__ y) {
    const int row  = blockIdx.x;
    const int lane = threadIdx.x;           // 0..31
    const float* xr = x + (size_t)row * D_;
    float v[16];
    float s = 0.f;
#pragma unroll
    for (int i = 0; i < 16; ++i) { v[i] = xr[lane * 16 + i]; s += v[i]; }
#pragma unroll
    for (int o = 16; o > 0; o >>= 1) s += __shfl_xor(s, o, 32);
    const float mean = s * (1.f / D_);
    float var = 0.f;
#pragma unroll
    for (int i = 0; i < 16; ++i) { float d = v[i] - mean; var += d * d; }
#pragma unroll
    for (int o = 16; o > 0; o >>= 1) var += __shfl_xor(var, o, 32);
    const float rstd = rsqrtf(var * (1.f / D_) + 1e-5f);
    __bf16* yr = y + (size_t)row * D_;
#pragma unroll
    for (int i = 0; i < 16; ++i) {
        int d = lane * 16 + i;
        yr[d] = (__bf16)((v[i] - mean) * rstd * g[d] + b[d]);
    }
}

// --------------------------------- FiLM ------------------------------------
__global__ void film_kernel(const float* __restrict__ cond,
                            const float* w0, const float* b0,
                            const float* w1, const float* b1,
                            const float* w2, const float* b2,
                            const float* w3, const float* b3,
                            float* __restrict__ films) {
    __shared__ float sc[D_];
    const int bb = blockIdx.x, f = blockIdx.y, t = threadIdx.x;
    for (int i = t; i < D_; i += 256) {
        float c = cond[(size_t)bb * D_ + i];
        sc[i] = c / (1.f + expf(-c));       // silu
    }
    __syncthreads();
    const float* w  = (f == 0) ? w0 : (f == 1) ? w1 : (f == 2) ? w2 : w3;
    const float* bi = (f == 0) ? b0 : (f == 1) ? b1 : (f == 2) ? b2 : b3;
    for (int o = t; o < 2 * D_; o += 256) {
        float acc = bi[o];
        const float* wr = w + (size_t)o * D_;
        for (int k = 0; k < D_; ++k) acc += sc[k] * wr[k];
        films[((size_t)f * B_ + bb) * (2 * D_) + o] = acc;
    }
}

// ---------------- bf16 WMMA GEMM: 64x64 block, LDS-staged ------------------
// Y[M,N] = A[M,K] @ W[N,K]^T + bias.  A,W bf16 row-major.
// grid = (ceil(M/64), N/64), block = 128 (4 waves, each a 32x32 subtile).
// Per 32-k chunk: stage A(64x32) + W(64x32) into double-buffered LDS via
// async global->LDS copies; fragments read back with ds_load_b128.
__device__ __forceinline__ void stage_tiles(const __bf16* A, int lda, int arow0,
                                            int Mrows,
                                            const __bf16* W, int ldw, int wrow0,
                                            __bf16* as, __bf16* ws, int kk,
                                            int tid) {
#pragma unroll
    for (int c = tid; c < 256; c += 128) {      // A: 64 rows x 4 chunks of 16B
        int row = c >> 2, ko = (c & 3) * 8;
        int gr = arow0 + row; if (gr >= Mrows) gr = Mrows - 1;
        copy16(A + (size_t)gr * lda + kk + ko, as + row * 32 + ko);
    }
#pragma unroll
    for (int c = tid; c < 256; c += 128) {      // W: 64 rows x 4 chunks of 16B
        int row = c >> 2, ko = (c & 3) * 8;
        copy16(W + (size_t)(wrow0 + row) * ldw + kk + ko, ws + row * 32 + ko);
    }
}

__global__ __launch_bounds__(128)
void gemm_wmma(const __bf16* __restrict__ A, int lda,
               const __bf16* __restrict__ W, int ldw,
               const float* __restrict__ bias,
               void* __restrict__ Yv, int ldy,
               int K, int Mrows, int flags) {
    __shared__ __align__(16) __bf16 As[2][64 * 32];
    __shared__ __align__(16) __bf16 Ws[2][64 * 32];
    const int tid  = threadIdx.x;
    const int wid  = tid >> 5, lane = tid & 31;
    const int wr   = wid >> 1, wc = wid & 1;      // wave -> 32x32 quadrant
    const int half = lane >> 4, nl = lane & 15;
    const int mt = blockIdx.x, nt = blockIdx.y;

    stage_tiles(A, lda, mt * 64, Mrows, W, ldw, nt * 64, As[0], Ws[0], 0, tid);

    v8f c00 = {}, c01 = {}, c10 = {}, c11 = {};
    const int nk = K >> 5;
    for (int t = 0; t < nk; ++t) {
        const int cur = t & 1;
        async_wait_all();
        __syncthreads();                 // staged buffer visible; prev drained
        if (t + 1 < nk)
            stage_tiles(A, lda, mt * 64, Mrows, W, ldw, nt * 64,
                        As[cur ^ 1], Ws[cur ^ 1], (t + 1) * 32, tid);
        const __bf16* as = As[cur];
        const __bf16* ws = Ws[cur];
        v16bf a0 = ldfrag(as + (wr * 32 + nl) * 32, half);
        v16bf a1 = ldfrag(as + (wr * 32 + 16 + nl) * 32, half);
        v16bf b0 = ldfrag(ws + (wc * 32 + nl) * 32, half);
        v16bf b1 = ldfrag(ws + (wc * 32 + 16 + nl) * 32, half);
        c00 = __builtin_amdgcn_wmma_f32_16x16x32_bf16(false, a0, false, b0,
                                                      (short)0, c00, false, false);
        c01 = __builtin_amdgcn_wmma_f32_16x16x32_bf16(false, a0, false, b1,
                                                      (short)0, c01, false, false);
        c10 = __builtin_amdgcn_wmma_f32_16x16x32_bf16(false, a1, false, b0,
                                                      (short)0, c10, false, false);
        c11 = __builtin_amdgcn_wmma_f32_16x16x32_bf16(false, a1, false, b1,
                                                      (short)0, c11, false, false);
    }

    float*  y32 = (float*)Yv;
    __bf16* y16 = (__bf16*)Yv;
    const int gelu = flags & FLAG_GELU, b16 = flags & FLAG_BF16OUT;
#pragma unroll
    for (int ij = 0; ij < 4; ++ij) {
        const int i = ij >> 1, j = ij & 1;
        const v8f& c = (ij == 0) ? c00 : (ij == 1) ? c01 : (ij == 2) ? c10 : c11;
        const int n = nt * 64 + wc * 32 + j * 16 + nl;
        const float bv = bias ? bias[n] : 0.f;
#pragma unroll
        for (int r = 0; r < 8; ++r) {
            int m = mt * 64 + wr * 32 + i * 16 + r + 8 * half;
            if (m >= Mrows) continue;
            float v = c[r] + bv;
            if (gelu) v = 0.5f * v * (1.f + erff(v * 0.70710678118654752f));
            if (b16) y16[(size_t)m * ldy + n] = (__bf16)v;
            else     y32[(size_t)m * ldy + n] = v;
        }
    }
}

// ------------------------------ attention ----------------------------------
// mode 0: temporal (seq=b*J+j, len T over stride J)
// mode 1: skeletal (seq=b*T+t, len J contiguous)
// mode 2: cross    (seq=b, TJ queries over M memory rows)
__device__ __forceinline__ long long tok_q(int mode, int seq, int l) {
    if (mode == 0) { int b = seq / J_, j = seq - b * J_; return ((long long)b * T_ + l) * J_ + j; }
    if (mode == 1) return (long long)seq * J_ + l;
    return (long long)seq * TJ_ + l;
}
__device__ __forceinline__ long long tok_kv(int mode, int seq, int s) {
    if (mode == 0) { int b = seq / J_, j = seq - b * J_; return ((long long)b * T_ + s) * J_ + j; }
    if (mode == 1) return (long long)seq * J_ + s;
    return (long long)seq * M_ + s;     // row into [B*M, ...] kv buffer
}
__device__ __forceinline__ size_t w_addr(int mode, int seq, int h, int l, int s) {
    if (mode == 0) return ((size_t)(seq * H_ + h)) * (T_ * T_) + (size_t)l * T_ + s;
    if (mode == 1) return ((size_t)(seq * H_ + h)) * (J_ * J_) + (size_t)l * J_ + s;
    return ((size_t)(seq * H_ + h)) * ((size_t)TJ_ * M_) + (size_t)l * M_ + s;
}

// grid = (ceil(L/16), H, nseq), block = 32 (one wave).
__global__ void attn_kernel(const __bf16* __restrict__ qbuf, int q_ld, int q_off,
                            const __bf16* __restrict__ kvbuf, int kv_ld,
                            int k_off, int v_off,
                            __bf16* __restrict__ ctx,       // [NTOK, 512] bf16
                            float* __restrict__ attnw,
                            int mode, int L, int S, int Spad32) {
    const int lane = threadIdx.x;
    const int lt = blockIdx.x, h = blockIdx.y, seq = blockIdx.z;
    const int half = lane >> 4, nl = lane & 15;
    __shared__ float sc[16 * 128];                 // 16 rows x Spad (<=128)

    // ---- scores = (Q K^T) * 1/sqrt(dh), tile by tile into LDS ----
    const int lq0 = lt * 16 + nl;
    const int lq  = (lq0 < L) ? lq0 : 0;
    const __bf16* qp = qbuf + tok_q(mode, seq, lq) * q_ld + q_off + h * DH_;
    const int stiles = (S + 15) / 16;
    for (int st = 0; st < stiles; ++st) {
        int s0 = st * 16 + nl;
        int s  = (s0 < S) ? s0 : 0;
        const __bf16* kp = kvbuf + tok_kv(mode, seq, s) * kv_ld + k_off + h * DH_;
        v8f c = {};
#pragma unroll
        for (int kk = 0; kk < DH_; kk += 32) {
            v16bf a = ldfrag(qp + kk, half);
            v16bf b = ldfrag(kp + kk, half);
            c = __builtin_amdgcn_wmma_f32_16x16x32_bf16(false, a, false, b,
                                                        (short)0, c, false, false);
        }
#pragma unroll
        for (int r = 0; r < 8; ++r) {
            int m = r + 8 * half;
            sc[m * 128 + st * 16 + nl] = c[r] * 0.125f;   // 1/sqrt(64)
        }
    }
    __syncthreads();

    // ---- softmax per row (exact, over valid S); pad cols -> prob 0 ----
    if (lane < 16) {
        const int m = lane;
        const int lrow = lt * 16 + m;
        float mx = -INFINITY;
        for (int s = 0; s < S; ++s) mx = fmaxf(mx, sc[m * 128 + s]);
        float sum = 0.f;
        for (int s = 0; s < S; ++s) {
            float e = expf(sc[m * 128 + s] - mx);
            sc[m * 128 + s] = e;
            sum += e;
        }
        const float inv = 1.f / sum;
        for (int s = 0; s < Spad32; ++s) {
            float p = (s < S) ? sc[m * 128 + s] * inv : 0.f;
            sc[m * 128 + s] = p;
            if (lrow < L && s < S) attnw[w_addr(mode, seq, h, lrow, s)] = p;
        }
    }
    __syncthreads();

    // ---- out = P @ V  (M=16, K=Spad32, N=dh in 4 tiles of 16) ----
#pragma unroll
    for (int vt = 0; vt < 4; ++vt) {
        v8f c = {};
        for (int kk = 0; kk < Spad32; kk += 32) {
            v16bf a, b;
#pragma unroll
            for (int e = 0; e < 16; ++e) {
                int k1 = kk + koff(e, half);
                a[e] = (__bf16)sc[nl * 128 + k1];
                int s = (k1 < S) ? k1 : 0;            // padded k has p==0
                b[e] = kvbuf[tok_kv(mode, seq, s) * kv_ld +
                             v_off + h * DH_ + vt * 16 + nl];
            }
            c = __builtin_amdgcn_wmma_f32_16x16x32_bf16(false, a, false, b,
                                                        (short)0, c, false, false);
        }
#pragma unroll
        for (int r = 0; r < 8; ++r) {
            int m = r + 8 * half;
            int lrow = lt * 16 + m;
            if (lrow < L)
                ctx[tok_q(mode, seq, lrow) * D_ + (size_t)h * DH_ + vt * 16 + nl] =
                    (__bf16)c[r];
        }
    }
}

// ------------------------ residual + FiLM affine ---------------------------
__global__ void add_affine(const float* __restrict__ xin,
                           const float* __restrict__ y,
                           const float* __restrict__ film,  // [B, 1024]
                           float* __restrict__ xout, size_t n) {
    size_t i = (size_t)blockIdx.x * blockDim.x + threadIdx.x;
    if (i >= n) return;
    int b = (int)(i / ((size_t)TJ_ * D_));
    int d = (int)(i & (D_ - 1));
    float scale = film[(size_t)b * (2 * D_) + d];
    float shift = film[(size_t)b * (2 * D_) + D_ + d];
    xout[i] = xin[i] + y[i] * (scale + 1.f) + shift;
}

// ---------------------------------------------------------------------------
extern "C" void kernel_launch(void* const* d_in, const int* in_sizes, int n_in,
                              void* d_out, int out_size, void* d_ws, size_t ws_size,
                              hipStream_t stream) {
    const float* x     = (const float*)d_in[0];
    const float* mem   = (const float*)d_in[1];
    const float* cond  = (const float*)d_in[2];
    const float* tqkvw = (const float*)d_in[3];
    const float* tqkvb = (const float*)d_in[4];
    const float* toutw = (const float*)d_in[5];
    const float* toutb = (const float*)d_in[6];
    const float* sqkvw = (const float*)d_in[7];
    const float* sqkvb = (const float*)d_in[8];
    const float* soutw = (const float*)d_in[9];
    const float* soutb = (const float*)d_in[10];
    const float* cqkvw = (const float*)d_in[11];
    const float* cqkvb = (const float*)d_in[12];
    const float* coutw = (const float*)d_in[13];
    const float* coutb = (const float*)d_in[14];
    const float* tlng  = (const float*)d_in[15];
    const float* tlnb  = (const float*)d_in[16];
    const float* slng  = (const float*)d_in[17];
    const float* slnb  = (const float*)d_in[18];
    const float* cslng = (const float*)d_in[19];
    const float* cslnb = (const float*)d_in[20];
    const float* ctlng = (const float*)d_in[21];
    const float* ctlnb = (const float*)d_in[22];
    const float* flng  = (const float*)d_in[23];
    const float* flnb  = (const float*)d_in[24];
    const float* ffw1  = (const float*)d_in[25];
    const float* ffb1  = (const float*)d_in[26];
    const float* ffw2  = (const float*)d_in[27];
    const float* ffb2  = (const float*)d_in[28];
    const float* fTw   = (const float*)d_in[29];
    const float* fTb   = (const float*)d_in[30];
    const float* fSw   = (const float*)d_in[31];
    const float* fSb   = (const float*)d_in[32];
    const float* fCw   = (const float*)d_in[33];
    const float* fCb   = (const float*)d_in[34];
    const float* fFw   = (const float*)d_in[35];
    const float* fFb   = (const float*)d_in[36];

    float* out = (float*)d_out;
    const size_t N = NTOK;
    float* out_x   = out;
    float* out_saw = out_x + N * D_;
    float* out_taw = out_saw + (size_t)B_ * T_ * H_ * J_ * J_;
    float* out_caw = out_taw + (size_t)B_ * J_ * H_ * T_ * T_;

    // -------- workspace carve-up (256B aligned) --------
    char* wsb = (char*)d_ws;
    auto carve = [&](size_t bytes) -> char* {
        char* p = wsb;
        wsb += (bytes + 255) & ~(size_t)255;
        return p;
    };
    float*  xcur  = (float*) carve(N * D_ * 4);            // running x (f32)
    float*  proj  = (float*) carve(N * D_ * 4);            // sublayer out (f32)
    __bf16* lnb   = (__bf16*)carve(N * D_ * 2);            // LN output
    __bf16* big   = (__bf16*)carve(N * (size_t)FF_ * 2);   // qkv / q+kv / ffn h1
    __bf16* ctx   = (__bf16*)carve(N * D_ * 2);            // attention context
    __bf16* memln = (__bf16*)carve((size_t)B_ * M_ * D_ * 2);
    float*  films = (float*) carve((size_t)4 * B_ * 2 * D_ * 4);
    // bf16 weights
    __bf16* wT  = (__bf16*)carve((size_t)3 * D_ * D_ * 2);
    __bf16* wS  = (__bf16*)carve((size_t)3 * D_ * D_ * 2);
    __bf16* wC  = (__bf16*)carve((size_t)3 * D_ * D_ * 2);
    __bf16* woT = (__bf16*)carve((size_t)D_ * D_ * 2);
    __bf16* woS = (__bf16*)carve((size_t)D_ * D_ * 2);
    __bf16* woC = (__bf16*)carve((size_t)D_ * D_ * 2);
    __bf16* wF1 = (__bf16*)carve((size_t)FF_ * D_ * 2);
    __bf16* wF2 = (__bf16*)carve((size_t)D_ * FF_ * 2);

    const int MT64 = (int)(N / 64);         // 720 row tiles
    const dim3 blk128(128), w32(32);

    (void)hipMemcpyAsync(xcur, x, N * D_ * sizeof(float),
                         hipMemcpyDeviceToDevice, stream);

    // weights -> bf16 (once per launch; ~10.5MB total, L2-resident after)
    cvt_bf16<<<2048, 256, 0, stream>>>(tqkvw, wT,  (size_t)3 * D_ * D_);
    cvt_bf16<<<2048, 256, 0, stream>>>(sqkvw, wS,  (size_t)3 * D_ * D_);
    cvt_bf16<<<2048, 256, 0, stream>>>(cqkvw, wC,  (size_t)3 * D_ * D_);
    cvt_bf16<<<1024, 256, 0, stream>>>(toutw, woT, (size_t)D_ * D_);
    cvt_bf16<<<1024, 256, 0, stream>>>(soutw, woS, (size_t)D_ * D_);
    cvt_bf16<<<1024, 256, 0, stream>>>(coutw, woC, (size_t)D_ * D_);
    cvt_bf16<<<2048, 256, 0, stream>>>(ffw1,  wF1, (size_t)FF_ * D_);
    cvt_bf16<<<2048, 256, 0, stream>>>(ffw2,  wF2, (size_t)D_ * FF_);

    film_kernel<<<dim3(B_, 4), 256, 0, stream>>>(cond, fTw, fTb, fSw, fSb,
                                                 fCw, fCb, fFw, fFb, films);
    float* filmT = films + 0 * (size_t)B_ * 2 * D_;
    float* filmS = films + 1 * (size_t)B_ * 2 * D_;
    float* filmC = films + 2 * (size_t)B_ * 2 * D_;
    float* filmF = films + 3 * (size_t)B_ * 2 * D_;

    const size_t nElem = N * D_;
    const int addBlocks = (int)((nElem + 255) / 256);

    // ---------------- temporal attention ----------------
    ln_kernel<<<(int)N, w32, 0, stream>>>(xcur, tlng, tlnb, lnb);
    gemm_wmma<<<dim3(MT64, (3 * D_) / 64), blk128, 0, stream>>>(
        lnb, D_, wT, D_, tqkvb, big, 3 * D_, D_, (int)N, FLAG_BF16OUT);
    attn_kernel<<<dim3((T_ + 15) / 16, H_, B_ * J_), w32, 0, stream>>>(
        big, 3 * D_, 0, big, 3 * D_, D_, 2 * D_, ctx, out_taw, 0, T_, T_, 128);
    gemm_wmma<<<dim3(MT64, D_ / 64), blk128, 0, stream>>>(
        ctx, D_, woT, D_, toutb, proj, D_, D_, (int)N, 0);
    add_affine<<<addBlocks, 256, 0, stream>>>(xcur, proj, filmT, xcur, nElem);

    // ---------------- skeletal attention ----------------
    ln_kernel<<<(int)N, w32, 0, stream>>>(xcur, slng, slnb, lnb);
    gemm_wmma<<<dim3(MT64, (3 * D_) / 64), blk128, 0, stream>>>(
        lnb, D_, wS, D_, sqkvb, big, 3 * D_, D_, (int)N, FLAG_BF16OUT);
    attn_kernel<<<dim3((J_ + 15) / 16, H_, B_ * T_), w32, 0, stream>>>(
        big, 3 * D_, 0, big, 3 * D_, D_, 2 * D_, ctx, out_saw, 1, J_, J_, 32);
    gemm_wmma<<<dim3(MT64, D_ / 64), blk128, 0, stream>>>(
        ctx, D_, woS, D_, soutb, proj, D_, D_, (int)N, 0);
    add_affine<<<addBlocks, 256, 0, stream>>>(xcur, proj, filmS, xcur, nElem);

    // ---------------- cross attention ----------------
    ln_kernel<<<(int)N, w32, 0, stream>>>(xcur, cslng, cslnb, lnb);
    ln_kernel<<<B_ * M_, w32, 0, stream>>>(mem, ctlng, ctlnb, memln);
    __bf16* qbuf  = big;                 // [N, 512] bf16
    __bf16* kvbuf = big + N * D_;        // [B*M, 1024] bf16
    gemm_wmma<<<dim3(MT64, D_ / 64), blk128, 0, stream>>>(
        lnb, D_, wC, D_, cqkvb, qbuf, D_, D_, (int)N, FLAG_BF16OUT);
    gemm_wmma<<<dim3((B_ * M_ + 63) / 64, (2 * D_) / 64), blk128, 0, stream>>>(
        memln, D_, wC + (size_t)D_ * D_, D_, cqkvb + D_, kvbuf, 2 * D_, D_,
        B_ * M_, FLAG_BF16OUT);
    attn_kernel<<<dim3(TJ_ / 16, H_, B_), w32, 0, stream>>>(
        qbuf, D_, 0, kvbuf, 2 * D_, 0, D_, ctx, out_caw, 2, TJ_, M_, 96);
    gemm_wmma<<<dim3(MT64, D_ / 64), blk128, 0, stream>>>(
        ctx, D_, woC, D_, coutb, proj, D_, D_, (int)N, 0);
    add_affine<<<addBlocks, 256, 0, stream>>>(xcur, proj, filmC, xcur, nElem);

    // ---------------- FFN ----------------
    ln_kernel<<<(int)N, w32, 0, stream>>>(xcur, flng, flnb, lnb);
    gemm_wmma<<<dim3(MT64, FF_ / 64), blk128, 0, stream>>>(
        lnb, D_, wF1, D_, ffb1, big, FF_, D_, (int)N, FLAG_GELU | FLAG_BF16OUT);
    gemm_wmma<<<dim3(MT64, D_ / 64), blk128, 0, stream>>>(
        big, FF_, wF2, FF_, ffb2, proj, D_, FF_, (int)N, 0);
    add_affine<<<addBlocks, 256, 0, stream>>>(xcur, proj, filmF, out_x, nElem);
}
